// Zone_82867099009081
// MI455X (gfx1250) — matrix operand order repaired and checked by reading, not compiled, not run on hardware
//
#include <hip/hip_runtime.h>
#include <hip/hip_bf16.h>
#include <math.h>
#include <stdint.h>

// ---------------------------------------------------------------------------
// new_states = states + 0.1 * gelu( einsum("bnd,nm->bmd", states, W*mask) + bias )
// B=8, N=4096, D=256.
//
// Two-pass plan (needs 48 MB workspace, else single-pass fallback):
//   pass 1a: Apk[m][k] = bf16(W[k,m]*mask[k,m])      (transpose+mask+cvt, once)
//   pass 1b: Spk[b][d][k] = bf16(states[b,k,d])      (transpose+cvt, once)
//   pass 2 : bf16 WMMA GEMM, double-buffered LDS, async global->LDS staging.
// ---------------------------------------------------------------------------

#define NN 4096
#define DD 256
#define NB 8

#define MT 128           // m-rows per block tile (pass 2)
#define KT 32            // k chunk per LDS stage (pass 2); row = 64 B = 4 x 16B chunks
#define NCHUNK (NN / KT) // 128
#define AROWS MT         // 128 rows in A tile
#define BROWS DD         // 256 rows in B tile
#define ABUF (AROWS * KT) // shorts per A buffer (4096)
#define BBUF (BROWS * KT) // shorts per B buffer (8192)

#define PPITCH 72        // pack kernels' LDS pitch (144 B, aligned + conflict-free)

typedef __bf16 v16bf __attribute__((ext_vector_type(16)));
typedef __bf16 v8bf  __attribute__((ext_vector_type(8)));
typedef float  v8f   __attribute__((ext_vector_type(8)));
typedef unsigned short v8us __attribute__((ext_vector_type(8)));
typedef int v4i __attribute__((vector_size(16)));
typedef __attribute__((address_space(1))) v4i gv4i;   // global int4
typedef __attribute__((address_space(3))) v4i lv4i;   // LDS int4

#if defined(__has_builtin)
#if __has_builtin(__builtin_amdgcn_global_load_async_to_lds_b128) && \
    __has_builtin(__builtin_amdgcn_s_wait_asynccnt)
#define USE_ASYNC 1
#endif
#endif

__device__ __forceinline__ unsigned short f32_to_bf16_rne(float f) {
    union { float f; unsigned int u; } v; v.f = f;
    unsigned int r = v.u + 0x7FFFu + ((v.u >> 16) & 1u);
    return (unsigned short)(r >> 16);
}

// 16-byte global -> LDS copy: async path (ASYNCcnt) if available, else load+store.
__device__ __forceinline__ void copy16(const unsigned short* g, unsigned short* l) {
#if USE_ASYNC
    __builtin_amdgcn_global_load_async_to_lds_b128(
        (gv4i*)(uintptr_t)g,                        // generic->as1: bit-identical
        (lv4i*)(unsigned int)(uintptr_t)l,          // generic->as3: low 32 bits
        0, 0);
#else
    *(v8us*)l = *(const v8us*)g;
#endif
}

__device__ __forceinline__ void wait_stage() {
#if USE_ASYNC
    __builtin_amdgcn_s_wait_asynccnt(0);
#endif
}

// ---------------- pass 1a: pack weights (mask fuse + transpose + bf16) -------
__global__ __launch_bounds__(256) void pack_weights(
    const float* __restrict__ W, const float* __restrict__ Cm,
    unsigned short* __restrict__ Apk)
{
    __shared__ unsigned short T[64 * PPITCH];
    const int tid = threadIdx.x;
    const int k0 = blockIdx.x * 64, m0 = blockIdx.y * 64;
    const int m4 = (tid & 15) << 2, kr = tid >> 4;
    #pragma unroll
    for (int j = 0; j < 4; ++j) {
        int k = kr + 16 * j;
        float4 w = *(const float4*)&W [(size_t)(k0 + k) * NN + m0 + m4];
        float4 c = *(const float4*)&Cm[(size_t)(k0 + k) * NN + m0 + m4];
        T[(m4 + 0) * PPITCH + k] = f32_to_bf16_rne(w.x * c.x);
        T[(m4 + 1) * PPITCH + k] = f32_to_bf16_rne(w.y * c.y);
        T[(m4 + 2) * PPITCH + k] = f32_to_bf16_rne(w.z * c.z);
        T[(m4 + 3) * PPITCH + k] = f32_to_bf16_rne(w.w * c.w);
    }
    __syncthreads();
    #pragma unroll
    for (int t = 0; t < 2; ++t) {
        int gl = tid + 256 * t, row = gl >> 3, kg = (gl & 7) << 3;
        *(v8us*)&Apk[(size_t)(m0 + row) * NN + k0 + kg] =
            *(const v8us*)&T[row * PPITCH + kg];
    }
}

// ---------------- pass 1b: pack states (transpose + bf16) --------------------
__global__ __launch_bounds__(256) void pack_states(
    const float* __restrict__ S, unsigned short* __restrict__ Spk)
{
    __shared__ unsigned short T[64 * PPITCH];
    const int tid = threadIdx.x;
    const int k0 = blockIdx.x * 64, d0 = blockIdx.y * 64, b = blockIdx.z;
    const int d4 = (tid & 15) << 2, kr = tid >> 4;
    #pragma unroll
    for (int j = 0; j < 4; ++j) {
        int k = kr + 16 * j;
        float4 s = *(const float4*)&S[((size_t)b * NN + k0 + k) * DD + d0 + d4];
        T[(d4 + 0) * PPITCH + k] = f32_to_bf16_rne(s.x);
        T[(d4 + 1) * PPITCH + k] = f32_to_bf16_rne(s.y);
        T[(d4 + 2) * PPITCH + k] = f32_to_bf16_rne(s.z);
        T[(d4 + 3) * PPITCH + k] = f32_to_bf16_rne(s.w);
    }
    __syncthreads();
    #pragma unroll
    for (int t = 0; t < 2; ++t) {
        int gl = tid + 256 * t, row = gl >> 3, kg = (gl & 7) << 3;
        *(v8us*)&Spk[((size_t)b * DD + d0 + row) * NN + k0 + kg] =
            *(const v8us*)&T[row * PPITCH + kg];
    }
}

// ---------------- pass 2: WMMA GEMM, double-buffered async staging -----------
// LDS tiles: rows of exactly 64 B (KT bf16), 16B-chunk XOR swizzle:
//   physical_chunk = logical_chunk ^ ((row>>2)&3)   -> aligned + bank-conflict-free
__global__ __launch_bounds__(256) void reservoir_gemm(
    const unsigned short* __restrict__ Apk,   // (N, N)  bf16, A=w_eff^T, k-contig
    const unsigned short* __restrict__ Spk,   // (B, D, N) bf16, k-contig
    const float* __restrict__ states,
    const float* __restrict__ bias,
    float* __restrict__ out)
{
    __shared__ unsigned short At[2 * ABUF];   // 16 KB
    __shared__ unsigned short Bt[2 * BBUF];   // 32 KB

    const int tid  = threadIdx.x;
    const int lane = tid & 31;
    const int wave = tid >> 5;      // 8 waves: 2 (m) x 4 (d)
    const int half = lane >> 4;
    const int l16  = lane & 15;
    const int wm   = wave & 1;
    const int wd   = wave >> 1;
    const int b    = blockIdx.z;
    const int m0   = blockIdx.y * MT;

    v8f acc[4][4] = {};

    // per-thread staging slots: gl -> (row, 16B chunk); A: 2 slots, B: 4 slots
    const unsigned short* aG[2]; int aL[2];
    const unsigned short* bG[4]; int bL[4];
    #pragma unroll
    for (int t = 0; t < 2; ++t) {
        int gl = tid + 256 * t, row = gl >> 2, kg = gl & 3;
        aG[t] = Apk + (size_t)(m0 + row) * NN + kg * 8;
        aL[t] = row * KT + ((kg ^ ((row >> 2) & 3)) << 3);
    }
    #pragma unroll
    for (int t = 0; t < 4; ++t) {
        int gl = tid + 256 * t, row = gl >> 2, kg = gl & 3;
        bG[t] = Spk + ((size_t)b * DD + row) * NN + kg * 8;
        bL[t] = row * KT + ((kg ^ ((row >> 2) & 3)) << 3);
    }

    // fragment LDS offsets (physical chunks after swizzle)
    int aOff[4][2], bOff[4][2];
    #pragma unroll
    for (int mi = 0; mi < 4; ++mi) {
        int row = wm * 64 + mi * 16 + l16, s = (row >> 2) & 3;
        aOff[mi][0] = row * KT + ((half ^ s) << 3);
        aOff[mi][1] = row * KT + (((2 + half) ^ s) << 3);
    }
    #pragma unroll
    for (int di = 0; di < 4; ++di) {
        int row = wd * 64 + di * 16 + l16, s = (row >> 2) & 3;
        bOff[di][0] = row * KT + ((half ^ s) << 3);
        bOff[di][1] = row * KT + (((2 + half) ^ s) << 3);
    }

    // prologue: stage chunk 0 into buffer 0
    #pragma unroll
    for (int t = 0; t < 2; ++t) copy16(aG[t], &At[aL[t]]);
    #pragma unroll
    for (int t = 0; t < 4; ++t) copy16(bG[t], &Bt[bL[t]]);
    wait_stage();
    __syncthreads();

    for (int i = 0; i < NCHUNK; ++i) {
        const int p = i & 1;
        // stage next chunk into other buffer (overlaps with WMMA below)
        if (i + 1 < NCHUNK) {
            const int kN = (i + 1) * KT, q = p ^ 1;
            #pragma unroll
            for (int t = 0; t < 2; ++t) copy16(aG[t] + kN, &At[q * ABUF + aL[t]]);
            #pragma unroll
            for (int t = 0; t < 4; ++t) copy16(bG[t] + kN, &Bt[q * BBUF + bL[t]]);
        }

        const unsigned short* Ac = &At[p * ABUF];
        const unsigned short* Bc = &Bt[p * BBUF];
        v16bf bfrag[4];
        #pragma unroll
        for (int di = 0; di < 4; ++di) {
            v16bf r;
            *((v8bf*)&r)     = *(const v8bf*)(Bc + bOff[di][0]);   // ds_load_b128
            *((v8bf*)&r + 1) = *(const v8bf*)(Bc + bOff[di][1]);   // ds_load_b128
            bfrag[di] = r;
        }
        #pragma unroll
        for (int mi = 0; mi < 4; ++mi) {
            v16bf a;
            *((v8bf*)&a)     = *(const v8bf*)(Ac + aOff[mi][0]);   // ds_load_b128
            *((v8bf*)&a + 1) = *(const v8bf*)(Ac + aOff[mi][1]);   // ds_load_b128
            #pragma unroll
            for (int di = 0; di < 4; ++di)
                acc[mi][di] = __builtin_amdgcn_wmma_f32_16x16x32_bf16(
                    false, a, false, bfrag[di], (short)0, acc[mi][di], false, false);
        }

        wait_stage();
        __syncthreads();
    }

    // epilogue: bias + exact-erf GELU + decayed residual (coalesced in d)
    #pragma unroll
    for (int mi = 0; mi < 4; ++mi) {
        #pragma unroll
        for (int di = 0; di < 4; ++di) {
            #pragma unroll
            for (int r = 0; r < 8; ++r) {
                int m = m0 + wm * 64 + mi * 16 + r + half * 8;
                int d = wd * 64 + di * 16 + l16;
                size_t oi = ((size_t)b * NN + m) * DD + d;
                float p = acc[mi][di][r] + bias[(size_t)m * DD + d];
                float g = 0.5f * p * (1.0f + erff(p * 0.70710678118654752440f));
                out[oi] = states[oi] + 0.1f * g;
            }
        }
    }
}

// ---------------- fallback: single-pass fused kernel (round-2 design) --------
#define FPITCH 72
__global__ __launch_bounds__(256) void reservoir_fused(
    const float* __restrict__ states, const float* __restrict__ W,
    const float* __restrict__ Cm, const float* __restrict__ bias,
    float* __restrict__ out)
{
    __shared__ unsigned short At[MT * FPITCH];
    __shared__ unsigned short Bt[DD * FPITCH];

    const int tid = threadIdx.x, lane = tid & 31, wave = tid >> 5;
    const int half = lane >> 4, l16 = lane & 15, wm = wave & 1, wd = wave >> 1;
    const int b = blockIdx.z, m0 = blockIdx.y * MT;

    v8f acc[4][4] = {};
    const int am = (tid & 31) << 2, ak = tid >> 5;
    const int bd = (tid & 63) << 2, bk = tid >> 6;
    const float* wBase = W + (size_t)m0 + am;
    const float* cBase = Cm + (size_t)m0 + am;
    const float* sBase = states + ((size_t)b * NN) * DD + bd;

    for (int k0 = 0; k0 < NN; k0 += 64) {
        __syncthreads();
        {
            const float4* wq = (const float4*)(wBase + (size_t)(k0 + ak) * NN);
            const float4* cq = (const float4*)(cBase + (size_t)(k0 + ak) * NN);
            #pragma unroll
            for (int j = 0; j < 8; ++j) {
                float4 w = wq[(size_t)j * 2 * NN];
                float4 c = cq[(size_t)j * 2 * NN];
                int k = ak + 8 * j;
                At[(am + 0) * FPITCH + k] = f32_to_bf16_rne(w.x * c.x);
                At[(am + 1) * FPITCH + k] = f32_to_bf16_rne(w.y * c.y);
                At[(am + 2) * FPITCH + k] = f32_to_bf16_rne(w.z * c.z);
                At[(am + 3) * FPITCH + k] = f32_to_bf16_rne(w.w * c.w);
            }
            const float4* sq = (const float4*)(sBase + (size_t)(k0 + bk) * DD);
            #pragma unroll
            for (int j = 0; j < 16; ++j) {
                float4 s = sq[(size_t)j * DD];
                int k = bk + 4 * j;
                Bt[(bd + 0) * FPITCH + k] = f32_to_bf16_rne(s.x);
                Bt[(bd + 1) * FPITCH + k] = f32_to_bf16_rne(s.y);
                Bt[(bd + 2) * FPITCH + k] = f32_to_bf16_rne(s.z);
                Bt[(bd + 3) * FPITCH + k] = f32_to_bf16_rne(s.w);
            }
        }
        __syncthreads();
        #pragma unroll
        for (int kk = 0; kk < 64; kk += 32) {
            v16bf bfrag[4];
            #pragma unroll
            for (int di = 0; di < 4; ++di) {
                const unsigned short* p =
                    &Bt[(wd * 64 + di * 16 + l16) * FPITCH + kk + half * 8];
                v16bf r;
                *((v8bf*)&r)     = *(const v8bf*)(p);
                *((v8bf*)&r + 1) = *(const v8bf*)(p + 16);
                bfrag[di] = r;
            }
            #pragma unroll
            for (int mi = 0; mi < 4; ++mi) {
                const unsigned short* p =
                    &At[(wm * 64 + mi * 16 + l16) * FPITCH + kk + half * 8];
                v16bf a;
                *((v8bf*)&a)     = *(const v8bf*)(p);
                *((v8bf*)&a + 1) = *(const v8bf*)(p + 16);
                #pragma unroll
                for (int di = 0; di < 4; ++di)
                    acc[mi][di] = __builtin_amdgcn_wmma_f32_16x16x32_bf16(
                        false, a, false, bfrag[di], (short)0, acc[mi][di], false, false);
            }
        }
    }
    #pragma unroll
    for (int mi = 0; mi < 4; ++mi)
        #pragma unroll
        for (int di = 0; di < 4; ++di)
            #pragma unroll
            for (int r = 0; r < 8; ++r) {
                int m = m0 + wm * 64 + mi * 16 + r + half * 8;
                int d = wd * 64 + di * 16 + l16;
                size_t oi = ((size_t)b * NN + m) * DD + d;
                float p = acc[mi][di][r] + bias[(size_t)m * DD + d];
                float g = 0.5f * p * (1.0f + erff(p * 0.70710678118654752440f));
                out[oi] = states[oi] + 0.1f * g;
            }
}

extern "C" void kernel_launch(void* const* d_in, const int* in_sizes, int n_in,
                              void* d_out, int out_size, void* d_ws, size_t ws_size,
                              hipStream_t stream) {
    const float* states = (const float*)d_in[0];
    const float* W      = (const float*)d_in[1];
    const float* Cm     = (const float*)d_in[2];
    const float* bias   = (const float*)d_in[3];
    float* out          = (float*)d_out;

    const size_t wsA = (size_t)NN * NN * 2;            // 32 MB bf16 packed weights
    const size_t wsS = (size_t)NB * DD * NN * 2;       // 16 MB bf16 packed states

    if (ws_size >= wsA + wsS) {
        unsigned short* Apk = (unsigned short*)d_ws;
        unsigned short* Spk = (unsigned short*)((char*)d_ws + wsA);
        pack_weights<<<dim3(NN / 64, NN / 64), 256, 0, stream>>>(W, Cm, Apk);
        pack_states <<<dim3(NN / 64, DD / 64, NB), 256, 0, stream>>>(states, Spk);
        reservoir_gemm<<<dim3(1, NN / MT, NB), 256, 0, stream>>>(
            Apk, Spk, states, bias, out);
    } else {
        reservoir_fused<<<dim3(1, NN / MT, NB), 256, 0, stream>>>(
            states, W, Cm, bias, out);
    }
}
